// OCS_5162550690084
// MI455X (gfx1250) — compile-verified, matrix operand order checked by reading, not run on hardware
//
#include <hip/hip_runtime.h>
#include <hip/hip_bf16.h>
#include <math.h>

#define BB 8
#define CC 128
#define HH 128
#define WW 128
#define LL (HH*WW)
#define MID 32
#define NBLK (BB * (LL / 32))  // 4096 tile-blocks in the fused GEMM kernels

typedef __attribute__((ext_vector_type(16))) __bf16 v16bf;
typedef __attribute__((ext_vector_type(8)))  float  v8f;

// ---------------------------------------------------------------------------
// CDNA5 WMMA 16-bit A/B fragment layout (wave32):
//   element e (0..15) of a lane's v16bf holds K = (e&7) + (e>=8?16:0) + (hi?8:0)
//   A: row M = lane&15 (both halves);  B: col N = lane&15.
//   C/D f32: VGPR r -> row r + 8*hi, col = lane&15.
// Inverse (k in 0..31 -> (hi,e)): hi=(k>>3)&1, e=(k&7)|(((k>>4)&1)<<3)
// Fragment-order storage keeps each (kk,hi,col) group of 16 elems contiguous,
// so A/B fragment loads are single 32B vector loads (b128 pairs).
// ---------------------------------------------------------------------------
__device__ __forceinline__ v8f wmma_bf16(v16bf a, v16bf b, v8f c) {
  return __builtin_amdgcn_wmma_f32_16x16x32_bf16(false, a, false, b, (short)0, c,
                                                 false, false);
}

__device__ __forceinline__ float elem32(v16bf a, v16bf b, int u) {
  return (u < 16) ? (float)a[u] : (float)b[u - 16];
}

// ---------------------------------------------------------------------------
// K0: pack weights to bf16 fragment order.
// mats 0..4: 128x128 -> dst[mat*16384 + ((o*4+kk)*2+hi)*16 + e]
// mat  5   : w_mlp2 128x32 -> dst[81920 + (o*2+hi)*16 + e]   (single kk)
// ---------------------------------------------------------------------------
__global__ __launch_bounds__(256) void k0_pack(const float* __restrict__ w0,
                                               const float* __restrict__ w1,
                                               const float* __restrict__ w2,
                                               const float* __restrict__ w3,
                                               const float* __restrict__ w4,
                                               const float* __restrict__ w5,
                                               __bf16* __restrict__ dst) {
  int flat = blockIdx.x * 256 + threadIdx.x;
  if (flat < 5 * CC * CC) {
    int mat = flat >> 14;
    int idx = flat & 16383;
    const float* src = (mat == 0) ? w0 : (mat == 1) ? w1 : (mat == 2) ? w2
                       : (mat == 3) ? w3 : w4;
    int o = idx >> 7, c = idx & 127;
    int kk = c >> 5, k = c & 31;
    int hi = (k >> 3) & 1;
    int e = (k & 7) | (((k >> 4) & 1) << 3);
    dst[mat * 16384 + (((o * 4 + kk) * 2 + hi) << 4) + e] = (__bf16)src[o * CC + c];
  } else if (flat < 5 * CC * CC + CC * MID) {
    int idx = flat - 5 * CC * CC;
    int o = idx >> 5, mm = idx & 31;
    int hi = (mm >> 3) & 1;
    int e = (mm & 7) | (((mm >> 4) & 1) << 3);
    dst[5 * CC * CC + ((o * 2 + hi) << 4) + e] = (__bf16)w5[o * MID + mm];
  }
}

// ---------------------------------------------------------------------------
// K1: per-(b,c) spatial mean of x -> gap
// ---------------------------------------------------------------------------
__global__ __launch_bounds__(256) void k1_gap(const float* __restrict__ x,
                                              float* __restrict__ gap) {
  __shared__ float red[256];
  int bc = blockIdx.x;
  int tid = threadIdx.x;
  const float4* p = (const float4*)(x + (size_t)bc * LL);
  float s = 0.f;
  for (int l4 = tid; l4 < LL / 4; l4 += 256) {
    float4 v = p[l4];
    s += v.x + v.y + v.z + v.w;
  }
  red[tid] = s;
  __syncthreads();
  for (int off = 128; off > 0; off >>= 1) {
    if (tid < off) red[tid] += red[tid + off];
    __syncthreads();
  }
  if (tid == 0) gap[bc] = red[0] * (1.f / (float)LL);
}

// ---------------------------------------------------------------------------
// K1b: normalize gap -> g; a = W_mlp1 @ g   (one block per batch)
// ---------------------------------------------------------------------------
__global__ __launch_bounds__(128) void k1b_prep(const float* __restrict__ gap,
                                                const float* __restrict__ wm1,
                                                float* __restrict__ g,
                                                float* __restrict__ a) {
  __shared__ float gs[CC];
  __shared__ float red[CC];
  int b = blockIdx.x;
  int c = threadIdx.x;
  float gv = gap[b * CC + c];
  red[c] = gv * gv;
  __syncthreads();
  for (int off = 64; off > 0; off >>= 1) {
    if (c < off) red[c] += red[c + off];
    __syncthreads();
  }
  float nrm = fmaxf(sqrtf(red[0]), 1e-12f);
  float gn = gv / nrm;
  gs[c] = gn;
  g[b * CC + c] = gn;
  __syncthreads();
  if (c < MID) {
    float s = 0.f;
#pragma unroll 8
    for (int k = 0; k < CC; ++k) s += wm1[c * CC + k] * gs[k];
    a[b * MID + c] = s;
  }
}

// ---------------------------------------------------------------------------
// K2: dual GEMM  U = W_sp_in @ x , V = W_ch_in @ x
// ---------------------------------------------------------------------------
__global__ __launch_bounds__(256) void k2_gemm_in(const float* __restrict__ x,
                                                  const __bf16* __restrict__ wpU,
                                                  const __bf16* __restrict__ wpV,
                                                  __bf16* __restrict__ U,
                                                  __bf16* __restrict__ V) {
  __shared__ __bf16 stage[4096];
  int blk = blockIdx.x;
  int b = blk >> 9;
  int lbase = (blk & 511) << 5;
  int tid = threadIdx.x;
  const float* xb = x + (size_t)b * CC * LL;
  {
    int c = tid >> 1, half = tid & 1;
    const float4* px = (const float4*)(xb + (size_t)c * LL + lbase + half * 16);
    float4 q0 = px[0], q1 = px[1], q2 = px[2], q3 = px[3];
    float qf[16] = {q0.x, q0.y, q0.z, q0.w, q1.x, q1.y, q1.z, q1.w,
                    q2.x, q2.y, q2.z, q2.w, q3.x, q3.y, q3.z, q3.w};
    int kk = c >> 5, k = c & 31;
    int hi = (k >> 3) & 1;
    int e = (k & 7) | (((k >> 4) & 1) << 3);
    int rb = ((kk * 2 + hi) * 32) * 16 + e;
#pragma unroll
    for (int u = 0; u < 16; ++u)
      stage[rb + (half * 16 + u) * 16] = (__bf16)qf[u];
  }
  __syncthreads();
  int wid = tid >> 5, lane = tid & 31;
  int m = lane & 15, hi = (lane >> 4) & 1;
  int obase = wid << 4;
  v8f au0 = {}, au1 = {}, av0 = {}, av1 = {};
#pragma unroll
  for (int kk = 0; kk < 4; ++kk) {
    v16bf aU = *(const v16bf*)(wpU + ((((obase + m) * 4 + kk) * 2 + hi) << 4));
    v16bf aV = *(const v16bf*)(wpV + ((((obase + m) * 4 + kk) * 2 + hi) << 4));
    v16bf b0 = *(const v16bf*)(&stage[(((kk * 2 + hi) * 32) + m) << 4]);
    v16bf b1 = *(const v16bf*)(&stage[(((kk * 2 + hi) * 32) + 16 + m) << 4]);
    au0 = wmma_bf16(aU, b0, au0);
    au1 = wmma_bf16(aU, b1, au1);
    av0 = wmma_bf16(aV, b0, av0);
    av1 = wmma_bf16(aV, b1, av1);
  }
#pragma unroll
  for (int r = 0; r < 8; ++r) {
    int o = obase + r + (hi << 3);
    size_t ro = ((size_t)(b * CC + o)) * LL + lbase;
    U[ro + m] = (__bf16)au0[r];
    U[ro + 16 + m] = (__bf16)au1[r];
    V[ro + m] = (__bf16)av0[r];
    V[ro + 16 + m] = (__bf16)av1[r];
  }
}

// ---------------------------------------------------------------------------
// K45: fully fused second half (stencil staging -> dual WMMA -> t -> silu MLP
// via WMMA -> z assembly with reflect-diff -> out-projection WMMA) + per-block
// per-channel BN partial sums via deterministic shfl_xor tree.
// ---------------------------------------------------------------------------
__global__ __launch_bounds__(256) void k45_fused(
    const __bf16* __restrict__ U, const __bf16* __restrict__ V,
    const __bf16* __restrict__ wpY, const __bf16* __restrict__ wpW,
    const __bf16* __restrict__ wp2, const __bf16* __restrict__ wpO,
    const float* __restrict__ g, const float* __restrict__ aP,
    const float* __restrict__ b1, const float* __restrict__ b2,
    const float* __restrict__ dw, const float* __restrict__ wdws,
    const float* __restrict__ wchd, const float* __restrict__ x,
    float* __restrict__ out, float* __restrict__ psum,
    float* __restrict__ psumsq) {
  __shared__ __bf16 stU[4096];  // Ud tile (frag order); later aliased as z tile
  __shared__ __bf16 stV[4096];  // Vd tile (frag order)
  __shared__ __bf16 h1f[2048];  // h1 (MID x 32 cols, frag order, single kk)
  __shared__ float part[2][256];
  __shared__ float tl[32];
  int blk = blockIdx.x;
  int b = blk >> 9;
  int lbase = (blk & 511) << 5;
  int tid = threadIdx.x;

  // ---- Phase A: fused stencil staging -------------------------------------
  {
    int s = tid >> 7, c = tid & 127;
    const __bf16* P = (s ? V : U) + ((size_t)(b * CC + c)) * LL;
    int y = lbase >> 7, xx0 = lbase & 127;
    v16bf q0 = *(const v16bf*)(P + lbase);
    v16bf q1 = *(const v16bf*)(P + lbase + 16);
    float lf0 = (lbase > 0) ? (float)P[lbase - 1] : 0.f;
    float rt31 = (lbase + 32 < LL) ? (float)P[lbase + 32] : 0.f;
    int kk = c >> 5, k = c & 31;
    int h2 = (k >> 3) & 1;
    int e = (k & 7) | (((k >> 4) & 1) << 3);
    int rb = ((kk * 2 + h2) * 32) * 16 + e;
    if (s) {  // channel conv
      float c0 = wchd[c * 3 + 0], c1 = wchd[c * 3 + 1], c2 = wchd[c * 3 + 2];
#pragma unroll
      for (int u = 0; u < 32; ++u) {
        float cu = elem32(q0, q1, u);
        float lf = (u > 0) ? elem32(q0, q1, u - 1) : lf0;
        float rt = (u < 31) ? elem32(q0, q1, u + 1) : rt31;
        stV[rb + (u << 4)] = (__bf16)(c0 * lf + c1 * cu + c2 * rt);
      }
    } else {  // spatial folded stencil
      float t1 = wdws[c * 3 + 1];
      float t02 = wdws[c * 3 + 0] + wdws[c * 3 + 2];
      bool yt = (y > 0), yb = (y < HH - 1);
      v16bf u0 = {}, u1 = {}, d0 = {}, d1 = {};
      if (yt) {
        u0 = *(const v16bf*)(P + lbase - WW);
        u1 = *(const v16bf*)(P + lbase - WW + 16);
      }
      if (yb) {
        d0 = *(const v16bf*)(P + lbase + WW);
        d1 = *(const v16bf*)(P + lbase + WW + 16);
      }
#pragma unroll
      for (int u = 0; u < 32; ++u) {
        int xx = xx0 + u;
        float cu = elem32(q0, q1, u);
        float lf = (u > 0) ? elem32(q0, q1, u - 1) : lf0;
        float rt = (u < 31) ? elem32(q0, q1, u + 1) : rt31;
        float up = yt ? elem32(u0, u1, u)
                      : ((xx > 0) ? (float)P[(HH - 1) * WW + xx - 1] : 0.f);
        float dn = yb ? elem32(d0, d1, u)
                      : ((xx < WW - 1) ? (float)P[xx + 1] : 0.f);
        stU[rb + (u << 4)] = (__bf16)(4.f * t1 * cu + t02 * (lf + rt + up + dn));
      }
    }
  }
  __syncthreads();

  // ---- Phase B: dual WMMA + t reduction -----------------------------------
  int wid = tid >> 5, lane = tid & 31;
  int m = lane & 15, hi = (lane >> 4) & 1;
  int obase = wid << 4;
  v8f ay0 = {}, ay1 = {}, aw0 = {}, aw1 = {};
#pragma unroll
  for (int kk = 0; kk < 4; ++kk) {
    v16bf aY = *(const v16bf*)(wpY + ((((obase + m) * 4 + kk) * 2 + hi) << 4));
    v16bf aW = *(const v16bf*)(wpW + ((((obase + m) * 4 + kk) * 2 + hi) << 4));
    v16bf bu0 = *(const v16bf*)(&stU[(((kk * 2 + hi) * 32) + m) << 4]);
    v16bf bu1 = *(const v16bf*)(&stU[(((kk * 2 + hi) * 32) + 16 + m) << 4]);
    v16bf bv0 = *(const v16bf*)(&stV[(((kk * 2 + hi) * 32) + m) << 4]);
    v16bf bv1 = *(const v16bf*)(&stV[(((kk * 2 + hi) * 32) + 16 + m) << 4]);
    ay0 = wmma_bf16(aY, bu0, ay0);
    ay1 = wmma_bf16(aY, bu1, ay1);
    aw0 = wmma_bf16(aW, bv0, aw0);
    aw1 = wmma_bf16(aW, bv1, aw1);
  }
  float p0 = 0.f, p1 = 0.f;
#pragma unroll
  for (int r = 0; r < 8; ++r) {
    float gv = g[b * CC + obase + r + (hi << 3)];
    p0 += gv * aw0[r];
    p1 += gv * aw1[r];
  }
  part[0][tid] = p0;
  part[1][tid] = p1;
  __syncthreads();
  if (tid < 32) {
    int gsel = tid >> 4, cl = tid & 15;
    float s = 0.f;
#pragma unroll
    for (int w = 0; w < 8; ++w)
      s += part[gsel][w * 32 + cl] + part[gsel][w * 32 + 16 + cl];
    tl[tid] = s;
  }
  __syncthreads();

  // ---- Phase C: h1 = silu(a*t + b1) into frag-order LDS -------------------
  for (int q = tid; q < MID * 32; q += 256) {
    int mm = q >> 5, col = q & 31;
    float hv = aP[b * MID + mm] * tl[col] + b1[mm];
    float sv = hv / (1.f + __expf(-hv));
    int h2 = (mm >> 3) & 1;
    int e = (mm & 7) | (((mm >> 4) & 1) << 3);
    h1f[((h2 * 32 + col) << 4) + e] = (__bf16)sv;
  }
  __syncthreads();

  // ---- Phase C2: ych via WMMA (K = MID = 32) ------------------------------
  v8f yc0 = {}, yc1 = {};
  {
    v16bf a2 = *(const v16bf*)(wp2 + (((obase + m) * 2 + hi) << 4));
    v16bf h0 = *(const v16bf*)(&h1f[((hi * 32) + m) << 4]);
    v16bf h1v = *(const v16bf*)(&h1f[((hi * 32) + 16 + m) << 4]);
    yc0 = wmma_bf16(a2, h0, yc0);
    yc1 = wmma_bf16(a2, h1v, yc1);
  }

  // ---- Phase D: assemble z into frag-order LDS (aliases stU) --------------
  float dwt = dw[0];
#pragma unroll
  for (int g2 = 0; g2 < 2; ++g2) {
    int col = g2 * 16 + m;
    int l = lbase + col;
    int y2 = l >> 7, xx = l & 127;
    int lx0 = (y2 << 7) + ((xx == 0) ? 1 : xx - 1);
    int lx1 = (y2 << 7) + ((xx == WW - 1) ? WW - 2 : xx + 1);
    int ly0 = (((y2 == 0) ? 1 : y2 - 1) << 7) + xx;
    int ly1 = (((y2 == HH - 1) ? HH - 2 : y2 + 1) << 7) + xx;
#pragma unroll
    for (int r = 0; r < 8; ++r) {
      int c = obase + r + (hi << 3);
      const float* xc = x + ((size_t)(b * CC + c)) * LL;
      float ctr = xc[l];
      float diff = 0.25f * (fabsf(ctr - xc[lx0]) + fabsf(ctr - xc[lx1]) +
                            fabsf(ctr - xc[ly0]) + fabsf(ctr - xc[ly1]));
      float ayv = g2 ? ay1[r] : ay0[r];
      float ycv = g2 ? yc1[r] : yc0[r];
      float z = 0.25f * ayv + ycv + b2[c] + ctr + dwt * diff;
      int kk = c >> 5, k = c & 31;
      int h2 = (k >> 3) & 1;
      int e = (k & 7) | (((k >> 4) & 1) << 3);
      stU[((((kk * 2 + h2) * 32) + col) << 4) + e] = (__bf16)z;
    }
  }
  __syncthreads();

  // ---- Phase E: out-projection WMMA, store pre-BN result ------------------
  v8f ac0 = {}, ac1 = {};
#pragma unroll
  for (int kk = 0; kk < 4; ++kk) {
    v16bf aO = *(const v16bf*)(wpO + ((((obase + m) * 4 + kk) * 2 + hi) << 4));
    v16bf z0 = *(const v16bf*)(&stU[(((kk * 2 + hi) * 32) + m) << 4]);
    v16bf z1 = *(const v16bf*)(&stU[(((kk * 2 + hi) * 32) + 16 + m) << 4]);
    ac0 = wmma_bf16(aO, z0, ac0);
    ac1 = wmma_bf16(aO, z1, ac1);
  }
#pragma unroll
  for (int r = 0; r < 8; ++r) {
    int o = obase + r + (hi << 3);
    size_t ro = ((size_t)(b * CC + o)) * LL + lbase;
    out[ro + m] = ac0[r];
    out[ro + 16 + m] = ac1[r];
  }

  // ---- Phase F: per-channel BN partials (deterministic shfl tree) ---------
#pragma unroll
  for (int r = 0; r < 8; ++r) {
    float sv = ac0[r] + ac1[r];
    float qv = ac0[r] * ac0[r] + ac1[r] * ac1[r];
#pragma unroll
    for (int mask = 1; mask < 16; mask <<= 1) {
      sv += __shfl_xor(sv, mask, 32);
      qv += __shfl_xor(qv, mask, 32);
    }
    if (m == 0) {  // lanes 0 and 16 hold their half's 32-col sums
      int o = obase + r + (hi << 3);
      psum[(size_t)o * NBLK + blk] = sv;
      psumsq[(size_t)o * NBLK + blk] = qv;
    }
  }
}

// ---------------------------------------------------------------------------
// K6a: reduce per-block partials -> per-channel mean / rstd (deterministic)
// ---------------------------------------------------------------------------
__global__ __launch_bounds__(256) void k6a_stats(const float* __restrict__ psum,
                                                 const float* __restrict__ psumsq,
                                                 float* __restrict__ mean,
                                                 float* __restrict__ rstd) {
  __shared__ float rs[256];
  __shared__ float rq[256];
  int c = blockIdx.x;
  int tid = threadIdx.x;
  float s = 0.f, q = 0.f;
  for (int i = tid; i < NBLK; i += 256) {
    s += psum[(size_t)c * NBLK + i];
    q += psumsq[(size_t)c * NBLK + i];
  }
  rs[tid] = s;
  rq[tid] = q;
  __syncthreads();
  for (int off = 128; off > 0; off >>= 1) {
    if (tid < off) {
      rs[tid] += rs[tid + off];
      rq[tid] += rq[tid + off];
    }
    __syncthreads();
  }
  if (tid == 0) {
    const float invN = 1.f / (float)(BB * LL);
    float mu = rs[0] * invN;
    float var = rq[0] * invN - mu * mu;
    mean[c] = mu;
    rstd[c] = rsqrtf(var + 1e-5f);
  }
}

// ---------------------------------------------------------------------------
// K6b: apply BN in place on d_out (float4)
// ---------------------------------------------------------------------------
__global__ __launch_bounds__(256) void k6b_apply(float* __restrict__ y,
                                                 const float* __restrict__ mean,
                                                 const float* __restrict__ rstd,
                                                 const float* __restrict__ gamma,
                                                 const float* __restrict__ beta) {
  size_t tot4 = (size_t)BB * CC * LL / 4;
  for (size_t i4 = (size_t)blockIdx.x * blockDim.x + threadIdx.x; i4 < tot4;
       i4 += (size_t)gridDim.x * blockDim.x) {
    int c = (int)((i4 * 4 / LL) % CC);
    float sc = rstd[c] * gamma[c];
    float mu = mean[c], bt = beta[c];
    float4* p = (float4*)y + i4;
    float4 v = *p;
    v.x = (v.x - mu) * sc + bt;
    v.y = (v.y - mu) * sc + bt;
    v.z = (v.z - mu) * sc + bt;
    v.w = (v.w - mu) * sc + bt;
    *p = v;
  }
}

// ---------------------------------------------------------------------------
extern "C" void kernel_launch(void* const* d_in, const int* in_sizes, int n_in,
                              void* d_out, int out_size, void* d_ws, size_t ws_size,
                              hipStream_t stream) {
  (void)in_sizes; (void)n_in; (void)out_size; (void)ws_size;
  const float* x    = (const float*)d_in[0];
  const float* wspi = (const float*)d_in[1];
  const float* wdws = (const float*)d_in[2];
  const float* wspo = (const float*)d_in[3];
  const float* wchi = (const float*)d_in[4];
  const float* wchd = (const float*)d_in[5];
  const float* wcho = (const float*)d_in[6];
  const float* wm1  = (const float*)d_in[7];
  const float* bm1  = (const float*)d_in[8];
  const float* wm2  = (const float*)d_in[9];
  const float* bm2  = (const float*)d_in[10];
  const float* dwt  = (const float*)d_in[11];
  const float* gam  = (const float*)d_in[12];
  const float* bet  = (const float*)d_in[13];
  const float* wop  = (const float*)d_in[14];
  float* out = (float*)d_out;

  const size_t N = (size_t)BB * CC * LL;  // 16,777,216 elements
  __bf16* bufA = (__bf16*)d_ws;           // U
  __bf16* bufB = bufA + N;                // V
  __bf16* wpack = bufB + N;               // 5*16384 + 4096 packed bf16 weights
  float* gapB = (float*)(wpack + 5 * CC * CC + CC * MID);
  float* gB = gapB + BB * CC;
  float* aB = gB + BB * CC;
  float* meanB = aB + BB * MID;
  float* rstdB = meanB + CC;
  float* psum = rstdB + CC;               // CC * NBLK
  float* psumsq = psum + (size_t)CC * NBLK;

  const __bf16* wpU = wpack + 0 * 16384;  // w_spatial_in
  const __bf16* wpV = wpack + 1 * 16384;  // w_ch_in
  const __bf16* wpY = wpack + 2 * 16384;  // w_spatial_out
  const __bf16* wpW = wpack + 3 * 16384;  // w_ch_out
  const __bf16* wpO = wpack + 4 * 16384;  // w_out_proj
  const __bf16* wp2 = wpack + 5 * 16384;  // w_mlp2 (128x32)

  k0_pack<<<(5 * CC * CC + CC * MID + 255) / 256, 256, 0, stream>>>(
      wspi, wchi, wspo, wcho, wop, wm2, wpack);
  k1_gap<<<BB * CC, 256, 0, stream>>>(x, gapB);
  k1b_prep<<<BB, 128, 0, stream>>>(gapB, wm1, gB, aB);
  k2_gemm_in<<<NBLK, 256, 0, stream>>>(x, wpU, wpV, bufA, bufB);
  k45_fused<<<NBLK, 256, 0, stream>>>(bufA, bufB, wpY, wpW, wp2, wpO, gB, aB,
                                      bm1, bm2, dwt, wdws, wchd, x, out, psum,
                                      psumsq);
  k6a_stats<<<CC, 256, 0, stream>>>(psum, psumsq, meanB, rstdB);
  k6b_apply<<<2048, 256, 0, stream>>>(out, meanB, rstdB, gam, bet);
}
